// ResGraphModule_11020886081778
// MI455X (gfx1250) — compile-verified, matrix operand order checked by reading, not run on hardware
//
#include <hip/hip_runtime.h>
#include <hip/hip_bf16.h>

// ---------------------------------------------------------------------------
// ResGraphModule fused pipeline for MI455X (gfx1250, wave32, WMMA)
//   e   = edge_attr @ W_edge              (bf16 WMMA, f32 accum)
//   agg = scatter_add(x[src] * e, dst)    (fused, global_atomic_add_f32)
//   h   = relu(agg@W_rel + b + x@W_root)  (bf16 WMMA)
//   out = BN(h)*gamma+beta + x@W_res      (bf16 WMMA for x@W_res)
//
// Memory-bound: x/agg (20MB each) live in 192MB L2 -> gathers + scatter
// atomics resolve in L2; edge_attr (82MB) streamed once from HBM.
// Weights are pre-swizzled into WMMA-fragment-major bf16 (32B/lane loads).
// Gathers/scatters use GVS addressing: uniform SGPR64 base + 32-bit VGPR
// byte offset + nt*64 immediate -> no per-iteration 64-bit address math.
// ---------------------------------------------------------------------------

typedef __attribute__((ext_vector_type(16))) __bf16 v16bf;
typedef __attribute__((ext_vector_type(8)))  float  v8f;

#define C_DIM 256   // IN == OUT
#define EC_DIM 64
#define TPW 4       // edge tiles per wave (amortize LDS fill)

static __device__ __forceinline__ unsigned short f2bfbits(float f) {
  unsigned u;
  __builtin_memcpy(&u, &f, 4);
  u += 0x7FFFu + ((u >> 16) & 1u);          // round-to-nearest-even
  return (unsigned short)(u >> 16);
}

static __device__ __forceinline__ unsigned pack2(float lo, float hi) {
  return (unsigned)f2bfbits(lo) | ((unsigned)f2bfbits(hi) << 16);
}

// K index for element e (0..15) of a 16-bit 16x32 A/B fragment, lane group g.
static __device__ __forceinline__ int kmap(int e, int g) {
  return (e < 8) ? (g * 8 + e) : (16 + g * 8 + (e - 8));
}

static __device__ __forceinline__ v8f wmma_bf16(v16bf a, v16bf b, v8f c) {
  return __builtin_amdgcn_wmma_f32_16x16x32_bf16(false, a, false, b,
                                                 (short)0, c, false, false);
}

// Build a 16x32 bf16 A fragment from a fp32 row. The kmap pattern per lane is
// two contiguous 8-float runs -> four float4 loads (b128) + packed convert.
static __device__ __forceinline__ v16bf load_a32(const float* __restrict__ row,
                                                 int kg) {
  const float4 q0 = ((const float4*)(row + kg * 8))[0];
  const float4 q1 = ((const float4*)(row + kg * 8))[1];
  const float4 q2 = ((const float4*)(row + 16 + kg * 8))[0];
  const float4 q3 = ((const float4*)(row + 16 + kg * 8))[1];
  union { v16bf v; unsigned u[8]; } t;
  t.u[0] = pack2(q0.x, q0.y); t.u[1] = pack2(q0.z, q0.w);
  t.u[2] = pack2(q1.x, q1.y); t.u[3] = pack2(q1.z, q1.w);
  t.u[4] = pack2(q2.x, q2.y); t.u[5] = pack2(q2.z, q2.w);
  t.u[6] = pack2(q3.x, q3.y); t.u[7] = pack2(q3.z, q3.w);
  return t.v;
}

// ---------------------------------------------------------------------------
// Prep kernels: swizzle fp32 weights into WMMA B-fragment-major bf16 images.
// 256x256 weight: dst[((nt*8+ks)*32+lane)*16+e] = bf16(W[k*256+n]),
//   k = ks*32+kmap(e,kg), n = nt*16+(lane&15), kg = lane>>4.   (65536 elems)
// ---------------------------------------------------------------------------
__global__ void __launch_bounds__(256)
swizzle_w256(const float* __restrict__ W, unsigned short* __restrict__ Wf) {
  const int d = blockIdx.x * 256 + threadIdx.x;        // 0 .. 65535
  const int e    = d & 15;
  const int lane = (d >> 4) & 31;
  const int ks   = (d >> 9) & 7;
  const int nt   = (d >> 12) & 15;
  const int kg = lane >> 4, rlane = lane & 15;
  const int k = ks * 32 + kmap(e, kg);
  const int n = nt * 16 + rlane;
  Wf[d] = f2bfbits(W[k * C_DIM + n]);
}

// 64x256 W_edge: dst[((nt*2+frag)*32+lane)*16+e], k = frag*32+kmap(e,kg).
__global__ void __launch_bounds__(256)
swizzle_wedge(const float* __restrict__ W, unsigned short* __restrict__ Wf) {
  const int d = blockIdx.x * 256 + threadIdx.x;        // 0 .. 16383
  const int e    = d & 15;
  const int lane = (d >> 4) & 31;
  const int frag = (d >> 9) & 1;
  const int nt   = (d >> 10) & 15;
  const int kg = lane >> 4, rlane = lane & 15;
  const int k = frag * 32 + kmap(e, kg);
  const int n = nt * 16 + rlane;
  Wf[d] = f2bfbits(W[k * C_DIM + n]);
}

// ---------------------------------------------------------------------------
// Kernel 1: edge phase. One wave = TPW 16-edge tiles.
//   e_tile[16x256] = edge_attr[16x64] @ W_edge[64x256]  (2 WMMA per N-tile)
//   agg[dst] += x[src] * e_tile                          (fp32 atomics -> L2)
// ---------------------------------------------------------------------------
__global__ void __launch_bounds__(256)
edge_kernel(const float* __restrict__ edge_attr,
            const int*   __restrict__ srcIdx,
            const int*   __restrict__ dstIdx,
            const float* __restrict__ x,
            const unsigned short* __restrict__ We_sw,   // swizzled bf16
            float*       __restrict__ agg,
            int E) {
  __shared__ unsigned short sWe[16 * 2 * 32 * 16];      // 32 KB fragment image
  {
    const uint4* g = (const uint4*)We_sw;
    uint4* s = (uint4*)sWe;
#pragma unroll
    for (int i = 0; i < 8; ++i)
      s[256 * i + threadIdx.x] = g[256 * i + threadIdx.x];
  }
  __syncthreads();

  const int lane  = threadIdx.x & 31;
  const int wave  = threadIdx.x >> 5;
  const int rlane = lane & 15;                 // row (A) / column (B,C,D)
  const int kg    = lane >> 4;                 // K-group; also C/D row half
  const int tile0 = (blockIdx.x * 8 + wave) * TPW;
  const unsigned short* fbase = sWe + lane * 16;  // + nt*2048(B0)/+1024(B1)

  const char* xb = (const char*)x;             // uniform SGPR64 bases
  char*       ab = (char*)agg;

#pragma unroll 1
  for (int t = 0; t < TPW; ++t) {
    const int base = (tile0 + t) * 16;
    if (base >= E) break;                      // wave-uniform

    // 32-bit VGPR byte offsets for the 8 result rows this lane sees
    // (M = r + 8*kg). Access pattern base + zext(off32) + nt*64 matches the
    // GVS global addressing mode (saddr + voffset + imm).
    unsigned xoff[8], aoff[8];
#pragma unroll
    for (int r = 0; r < 8; ++r) {
      const int m = base + r + 8 * kg;
      xoff[r] = (unsigned)srcIdx[m] * (C_DIM * 4u) + (unsigned)rlane * 4u;
      aoff[r] = (unsigned)dstIdx[m] * (C_DIM * 4u) + (unsigned)rlane * 4u;
    }

    const float* earow = edge_attr + (size_t)(base + rlane) * EC_DIM;
    const v16bf a0 = load_a32(earow, kg);      // K =  0..31
    const v16bf a1 = load_a32(earow + 32, kg); // K = 32..63

#pragma unroll
    for (int nt = 0; nt < 16; ++nt) {
      const v16bf b0 = *(const v16bf*)(fbase + nt * 1024);        // ds b128 x2
      const v16bf b1 = *(const v16bf*)(fbase + nt * 1024 + 512);
      v8f c = {0.f, 0.f, 0.f, 0.f, 0.f, 0.f, 0.f, 0.f};
      c = wmma_bf16(a0, b0, c);
      c = wmma_bf16(a1, b1, c);

      const unsigned imm = (unsigned)nt * 64u; // constant after unroll
      float xg[8];
#pragma unroll
      for (int r = 0; r < 8; ++r)              // 8 gathers in one clause
        xg[r] = *(const float*)(xb + ((size_t)xoff[r] + imm));
#pragma unroll
      for (int r = 0; r < 8; ++r) {
        float* p = (float*)(ab + ((size_t)aoff[r] + imm));
        __hip_atomic_fetch_add(p, c[r] * xg[r],
                               __ATOMIC_RELAXED, __HIP_MEMORY_SCOPE_AGENT);
      }
    }
  }
}

// ---------------------------------------------------------------------------
// Kernel 2: node phase. One wave = one 16-node tile.
//   accH = agg@W_rel + x@W_root ; accR = x@W_res   (24 WMMA per N-tile)
//   h = relu(accH + b) -> hbuf ; accR -> out ; BN partial sums via atomics
// ---------------------------------------------------------------------------
__global__ void __launch_bounds__(256)
node_kernel(const float* __restrict__ agg,
            const float* __restrict__ x,
            const unsigned short* __restrict__ Wrel_sw,
            const unsigned short* __restrict__ Wroot_sw,
            const unsigned short* __restrict__ Wres_sw,
            const float* __restrict__ b_rel,
            float* __restrict__ hbuf,
            float* __restrict__ outbuf,
            float* __restrict__ bn_sum,
            float* __restrict__ bn_ssq,
            int N) {
  const int lane  = threadIdx.x & 31;
  const int wave  = threadIdx.x >> 5;
  const int tile  = blockIdx.x * 8 + wave;
  const int base  = tile * 16;
  if (base >= N) return;

  const int rlane = lane & 15;
  const int kg    = lane >> 4;
  const int row   = base + rlane;

  // Full K=256 A panels (agg and x rows) as bf16 fragments, held in VGPRs.
  v16bf aA[8], aX[8];
  const float* arow = agg + (size_t)row * C_DIM;
  const float* xrow = x   + (size_t)row * C_DIM;
#pragma unroll
  for (int ks = 0; ks < 8; ++ks) {
    aA[ks] = load_a32(arow + ks * 32, kg);
    aX[ks] = load_a32(xrow + ks * 32, kg);
  }

  const v16bf* relF  = (const v16bf*)Wrel_sw;
  const v16bf* rootF = (const v16bf*)Wroot_sw;
  const v16bf* resF  = (const v16bf*)Wres_sw;

  float* hbase = hbuf   + (size_t)(base + 8 * kg) * C_DIM + rlane;
  float* obase = outbuf + (size_t)(base + 8 * kg) * C_DIM + rlane;

#pragma unroll 1
  for (int nt = 0; nt < 16; ++nt) {
    if (nt < 15) {                              // global_prefetch_b8
      __builtin_prefetch(&relF [(nt + 1) * 256 + lane], 0, 1);
      __builtin_prefetch(&rootF[(nt + 1) * 256 + lane], 0, 1);
      __builtin_prefetch(&resF [(nt + 1) * 256 + lane], 0, 1);
    }
    v8f accH = {0.f, 0.f, 0.f, 0.f, 0.f, 0.f, 0.f, 0.f};
    v8f accR = {0.f, 0.f, 0.f, 0.f, 0.f, 0.f, 0.f, 0.f};
#pragma unroll
    for (int ks = 0; ks < 8; ++ks) {
      const int fi = nt * 256 + ks * 32 + lane; // (nt*8+ks)*32+lane
      const v16bf bRel  = relF [fi];            // one 32B coalesced load each
      const v16bf bRoot = rootF[fi];
      const v16bf bRes  = resF [fi];
      accH = wmma_bf16(aA[ks], bRel,  accH);
      accH = wmma_bf16(aX[ks], bRoot, accH);
      accR = wmma_bf16(aX[ks], bRes,  accR);
    }

    const int n = nt * 16 + rlane;
    const float bias = b_rel[n];
    float psum = 0.f, psq = 0.f;
    float* hp = hbase + nt * 16;
    float* op = obase + nt * 16;
#pragma unroll
    for (int r = 0; r < 8; ++r) {
      const float h = fmaxf(accH[r] + bias, 0.f);
      hp[r * C_DIM] = h;
      op[r * C_DIM] = accR[r];                  // residual branch x@W_res
      psum += h;
      psq  += h * h;
    }
    __hip_atomic_fetch_add(&bn_sum[n], psum,
                           __ATOMIC_RELAXED, __HIP_MEMORY_SCOPE_AGENT);
    __hip_atomic_fetch_add(&bn_ssq[n], psq,
                           __ATOMIC_RELAXED, __HIP_MEMORY_SCOPE_AGENT);
  }
}

// ---------------------------------------------------------------------------
// Kernel 3: fold BN stats into per-channel scale/shift.
// ---------------------------------------------------------------------------
__global__ void bn_finalize(const float* __restrict__ bn_sum,
                            const float* __restrict__ bn_ssq,
                            const float* __restrict__ gamma,
                            const float* __restrict__ beta,
                            float* __restrict__ scale,
                            float* __restrict__ shift,
                            int N) {
  const int ch = threadIdx.x;
  if (ch < C_DIM) {
    const float invN = 1.0f / (float)N;
    const float mean = bn_sum[ch] * invN;
    const float var  = bn_ssq[ch] * invN - mean * mean;
    const float inv  = rsqrtf(var + 1e-5f);
    const float sc   = inv * gamma[ch];
    scale[ch] = sc;
    shift[ch] = beta[ch] - mean * sc;
  }
}

// ---------------------------------------------------------------------------
// Kernel 4: out = h*scale + shift + out  (out already holds x@W_res)
// ---------------------------------------------------------------------------
__global__ void __launch_bounds__(256)
apply_kernel(const float* __restrict__ hbuf,
             const float* __restrict__ scale,
             const float* __restrict__ shift,
             float* __restrict__ out,
             int total4) {
  const int i = blockIdx.x * 256 + threadIdx.x;
  if (i >= total4) return;
  const float4 h4 = ((const float4*)hbuf)[i];
  float4 o4 = ((float4*)out)[i];
  const int ch = (i * 4) & (C_DIM - 1);
  o4.x = h4.x * scale[ch + 0] + shift[ch + 0] + o4.x;
  o4.y = h4.y * scale[ch + 1] + shift[ch + 1] + o4.y;
  o4.z = h4.z * scale[ch + 2] + shift[ch + 2] + o4.z;
  o4.w = h4.w * scale[ch + 3] + shift[ch + 3] + o4.w;
  ((float4*)out)[i] = o4;
}

// ---------------------------------------------------------------------------
// Host-side launcher (graph-capture safe: only async ops on `stream`).
// Inputs: x, edge_index, edge_attr, W_edge, W_rel, b_rel, W_root, gamma,
//         beta, W_res
// ---------------------------------------------------------------------------
extern "C" void kernel_launch(void* const* d_in, const int* in_sizes, int n_in,
                              void* d_out, int out_size, void* d_ws, size_t ws_size,
                              hipStream_t stream) {
  const float* x         = (const float*)d_in[0];
  const int*   edge_idx  = (const int*)  d_in[1];
  const float* edge_attr = (const float*)d_in[2];
  const float* W_edge    = (const float*)d_in[3];
  const float* W_rel     = (const float*)d_in[4];
  const float* b_rel     = (const float*)d_in[5];
  const float* W_root    = (const float*)d_in[6];
  const float* gamma     = (const float*)d_in[7];
  const float* beta      = (const float*)d_in[8];
  const float* W_res     = (const float*)d_in[9];
  float*       out       = (float*)d_out;

  const int N = in_sizes[0] / C_DIM;   // 20000 (multiple of 16)
  const int E = in_sizes[1] / 2;       // 320000 (multiple of 16)
  const int* src = edge_idx;
  const int* dst = edge_idx + E;

  // Workspace layout (floats, then bf16 images; all 32B-aligned):
  //   agg[N*256] | h[N*256] | sum[256] ssq[256] scale[256] shift[256]
  //   | We_sw[16384] | Wrel_sw[65536] | Wroot_sw[65536] | Wres_sw[65536]
  float* ws    = (float*)d_ws;
  float* agg   = ws;
  float* hbuf  = agg + (size_t)N * C_DIM;
  float* sum   = hbuf + (size_t)N * C_DIM;
  float* ssq   = sum + C_DIM;
  float* scale = ssq + C_DIM;
  float* shift = scale + C_DIM;
  unsigned short* We_sw    = (unsigned short*)(shift + C_DIM);
  unsigned short* Wrel_sw  = We_sw + 16384;
  unsigned short* Wroot_sw = Wrel_sw + 65536;
  unsigned short* Wres_sw  = Wroot_sw + 65536;

  hipMemsetAsync(agg, 0, (size_t)N * C_DIM * sizeof(float), stream);
  hipMemsetAsync(sum, 0, 2 * C_DIM * sizeof(float), stream);   // sum+ssq

  // Weight pre-swizzle (cheap: ~416KB of writes, fully deterministic).
  swizzle_wedge<<<16384 / 256, 256, 0, stream>>>(W_edge, We_sw);
  swizzle_w256 <<<65536 / 256, 256, 0, stream>>>(W_rel,  Wrel_sw);
  swizzle_w256 <<<65536 / 256, 256, 0, stream>>>(W_root, Wroot_sw);
  swizzle_w256 <<<65536 / 256, 256, 0, stream>>>(W_res,  Wres_sw);

  const int etiles = (E + 15) / 16;
  edge_kernel<<<(etiles + 8 * TPW - 1) / (8 * TPW), 256, 0, stream>>>(
      edge_attr, src, dst, x, We_sw, agg, E);

  const int ntiles = (N + 15) / 16;
  node_kernel<<<(ntiles + 7) / 8, 256, 0, stream>>>(
      agg, x, Wrel_sw, Wroot_sw, Wres_sw, b_rel, hbuf, out, sum, ssq, N);

  bn_finalize<<<1, 256, 0, stream>>>(sum, ssq, gamma, beta, scale, shift, N);

  const int total4 = N * C_DIM / 4;
  apply_kernel<<<(total4 + 255) / 256, 256, 0, stream>>>(
      hbuf, scale, shift, out, total4);
}
